// MAB_sb_7198365188349
// MI455X (gfx1250) — compile-verified, matrix operand order checked by reading, not run on hardware
//
#include <hip/hip_runtime.h>
#include <hip/hip_bf16.h>

// Problem constants (from reference)
#define DV   1024
#define HN   16
#define DH   64
#define BN   4
#define NQ   1024
#define NK   1024
#define MROWS 4096   // B*NQ

// LDS pitch for TDM-staged weight tiles: 16 dwords/row + 1 pad dword per 8
// dwords -> 18 dwords = 36 bf16 = 18 banks (conflict-free for 16-lane reads).
#define WPITCH 36
#define WBUFSZ (128 * WPITCH)   // bf16 elements per buffer

// ---------------------------------------------------------------- WMMA types
typedef __attribute__((ext_vector_type(16))) __bf16 v16bf;
typedef __attribute__((ext_vector_type(8)))  float  v8f;
typedef __attribute__((ext_vector_type(4)))  unsigned int v4ui;
typedef __attribute__((ext_vector_type(8)))  int v8si;
typedef __attribute__((ext_vector_type(4)))  int v4si;

union Frag {
  v16bf bf;
  unsigned int u[8];
};

__device__ __forceinline__ unsigned short f2bf(float f) {
  unsigned u = __builtin_bit_cast(unsigned, f);
  u += 0x7FFFu + ((u >> 16) & 1u);          // round-to-nearest-even
  return (unsigned short)(u >> 16);
}

__device__ __forceinline__ unsigned pack_bf2(float lo, float hi) {
  return (unsigned)f2bf(lo) | ((unsigned)f2bf(hi) << 16);
}

__device__ __forceinline__ v8f wmma_bf16(const Frag& a, const Frag& b, v8f c) {
  return __builtin_amdgcn_wmma_f32_16x16x32_bf16(
      /*neg_a=*/false, a.bf, /*neg_b=*/false, b.bf,
      /*c_mod=*/(short)0, c, /*reuse_a=*/false, /*reuse_b=*/false);
}

// Load a 16x32 bf16 A/B fragment from row-major memory (ld = row pitch in
// elements, must be even). Layout per CDNA5 ISA 7.12.2:
//   lane r (r=lane&15) owns row/col r; lanes 0-15 hold K={0..7,16..23},
//   lanes 16-31 hold K={8..15,24..31}; 2 consecutive K per VGPR.
// hi_off = element offset of K=16 (16 for dense rows, 18 for TDM-padded LDS).
template <typename P>
__device__ __forceinline__ Frag load_frag(const P* p, int ld, int hi_off = 16) {
  Frag f;
  const int lane = threadIdx.x & 31;
  const int r = lane & 15;
  const int h = lane >> 4;
  const unsigned int* lo = (const unsigned int*)(p + (size_t)r * ld + h * 8);
  const unsigned int* hi = (const unsigned int*)(p + (size_t)r * ld + hi_off + h * 8);
#pragma unroll
  for (int j = 0; j < 4; ++j) { f.u[j] = lo[j]; f.u[4 + j] = hi[j]; }
  return f;
}

// -------------------------------------------------- Tensor Data Mover stage
// DMA a 128-row x 16-dword (32 bf16) weight slab (row stride 512 dwords) into
// LDS, padding 1 dword after every 8 dwords (LDS row pitch 18 dwords).
// One issue per calling wave; completion tracked by TENSORcnt.
__device__ __forceinline__ void tdm_stage_weights(const unsigned short* gsrc,
                                                  unsigned lds_byte_off) {
  const unsigned long long ga = (unsigned long long)(size_t)gsrc;
  v4ui g0;
  g0[0] = 0x1u;                                   // count=1, user descriptor
  g0[1] = lds_byte_off;                           // lds_addr (bytes)
  g0[2] = (unsigned)(ga & 0xFFFFFFFFu);           // global_addr[31:0]
  g0[3] = (unsigned)((ga >> 32) & 0x1FFFFFFu)     // global_addr[56:32]
        | (2u << 30);                             // type = 2 ("image")
  v8si g1;
  g1[0] = (int)((2u << 16)                        // data_size = 4 bytes
              | (1u << 20)                        // pad_enable
              | (2u << 22)                        // pad_interval: 8 dwords
              | (0u << 25));                      // pad_amount: 1 dword
  g1[1] = (int)((512u & 0xFFFFu) << 16);          // tensor_dim0[15:0] (512 dw/row)
  g1[2] = (int)((512u >> 16) | ((1024u & 0xFFFFu) << 16)); // td0 hi | tensor_dim1 lo
  g1[3] = (int)((1024u >> 16) | (16u << 16));     // td1 hi | tile_dim0 = 16 dwords
  g1[4] = (int)128u;                              // tile_dim1 = 128 rows, tile_dim2 = 0
  g1[5] = (int)512u;                              // tensor_dim0_stride[31:0] = 512 dwords
  g1[6] = 0;                                      // stride hi | dim1_stride lo (unused, 2D)
  g1[7] = 0;
  const v4si z4 = {0, 0, 0, 0};                   // groups 2/3 unused (<=2D tensor)
  const v8si z8 = {0, 0, 0, 0, 0, 0, 0, 0};
  __builtin_amdgcn_tensor_load_to_lds(g0, g1, z4, z4, z8, 0);
}

// ------------------------------------------------------------- f32 -> bf16
__global__ __launch_bounds__(256)
void cvt_bf16_kernel(const float4* __restrict__ in, uint2* __restrict__ out, int n4) {
  int i = blockIdx.x * 256 + threadIdx.x;
  if (i < n4) {
    float4 v = in[i];
    out[i] = make_uint2(pack_bf2(v.x, v.y), pack_bf2(v.z, v.w));
  }
}

// --------------------------------------------------- fused QKV projections
// grid (32, 8, 3); block 256 = 8 waves. Block tile 128x128, wave tile 16x128.
// Weight slabs streamed into double-buffered LDS by the TDM, overlapped with
// WMMA on the previous slab.
// z=0: q = Q@Wq^T+bq  -> qf (f32) + qb (bf16), head-major [h,b,i,d]
// z=1: k = K@Wk^T+bk  -> khb bf16 [h,b,i,d]   (B-matrix layout for Q*K^T)
// z=2: v = K@Wv^T+bv  -> vtb bf16 [h,b,d,i]   (B-matrix layout for P*V)
__global__ __launch_bounds__(256)
void qkv_gemm_kernel(const unsigned short* __restrict__ Qb,
                     const unsigned short* __restrict__ Kb,
                     const unsigned short* __restrict__ Wqb,
                     const unsigned short* __restrict__ Wkb,
                     const unsigned short* __restrict__ Wvb,
                     const float* __restrict__ bq,
                     const float* __restrict__ bk,
                     const float* __restrict__ bv,
                     float* __restrict__ qf,
                     unsigned short* __restrict__ qb,
                     unsigned short* __restrict__ khb,
                     unsigned short* __restrict__ vtb) {
  __shared__ __align__(16) unsigned short ldsB[2 * WBUFSZ];
  const int z = blockIdx.z;
  const unsigned short* A = (z == 0) ? Qb : Kb;
  const unsigned short* W = (z == 0) ? Wqb : (z == 1) ? Wkb : Wvb;
  const float* bias = (z == 0) ? bq : (z == 1) ? bk : bv;

  const int wave  = threadIdx.x >> 5;
  const int lane  = threadIdx.x & 31;
  const int cn    = lane & 15;
  const int mh    = (lane >> 4) << 3;
  const int mtile = blockIdx.x * 128 + wave * 16;
  const int ntile = blockIdx.y * 128;
  const bool issuer = (wave == 0);

  v8f acc[8] = {};

  const unsigned ldsbase = (unsigned)(size_t)&ldsB[0];
  const int NIT = DV / 32;

  if (issuer) tdm_stage_weights(W + (size_t)ntile * DV, ldsbase);

  for (int it = 0; it < NIT; ++it) {
    const int k0 = it * 32;
    if (it + 1 < NIT) {
      if (issuer) {
        tdm_stage_weights(W + (size_t)ntile * DV + (k0 + 32),
                          ldsbase + ((it + 1) & 1) * (WBUFSZ * 2));
        __builtin_amdgcn_s_wait_tensorcnt(1);   // slab 'it' landed
      }
    } else if (issuer) {
      __builtin_amdgcn_s_wait_tensorcnt(0);
    }
    __syncthreads();

    const unsigned short* bufp = ldsB + (it & 1) * WBUFSZ;
    Frag af = load_frag(A + (size_t)mtile * DV + k0, DV);
#pragma unroll
    for (int nb = 0; nb < 8; ++nb) {
      Frag bf = load_frag(bufp + nb * 16 * WPITCH, WPITCH, 18);
      acc[nb] = wmma_bf16(af, bf, acc[nb]);
    }
    __syncthreads();
  }

  // epilogue: bias add + head split
#pragma unroll
  for (int nb = 0; nb < 8; ++nb) {
    const int n = ntile + nb * 16 + cn;
    const int hh = n >> 6, d = n & 63;
    const float bias_n = bias[n];
#pragma unroll
    for (int v = 0; v < 8; ++v) {
      const int m = mtile + mh + v;
      const int bb = m >> 10, i = m & 1023;
      const float val = acc[nb][v] + bias_n;
      if (z == 0) {
        const size_t o = ((size_t)(hh * BN + bb) * NQ + i) * DH + d;
        qf[o] = val;
        qb[o] = f2bf(val);
      } else if (z == 1) {
        khb[((size_t)(hh * BN + bb) * NK + i) * DH + d] = f2bf(val);
      } else {
        vtb[((size_t)(hh * BN + bb) * DH + d) * NK + i] = f2bf(val);
      }
    }
  }
}

// --------------------------------------------------------- flash attention
// grid (64, B, H); block 32 (one wave). 16 query rows per wave, 64 keys/iter.
// oh = q + softmax(q*k^T / 32 + bias) * v   (scale = 1/sqrt(DV))
__global__ __launch_bounds__(32)
void attn_kernel(const unsigned short* __restrict__ qb,
                 const unsigned short* __restrict__ khb,
                 const unsigned short* __restrict__ vtb,
                 const float* __restrict__ qf,
                 const float* __restrict__ sbias,
                 float* __restrict__ OH) {
  __shared__ __align__(16) unsigned short Plds[16 * 66];  // 66-pitch: no conflicts
  const int h = blockIdx.z, b = blockIdx.y, q0 = blockIdx.x * 16;
  const int lane = threadIdx.x & 31;
  const int cn = lane & 15;
  const int mh = (lane >> 4) << 3;

  const size_t hb = (size_t)(h * BN + b);
  const unsigned short* qp = qb + (hb * NQ + q0) * DH;
  const unsigned short* kp = khb + hb * NK * DH;
  const unsigned short* vp = vtb + hb * (size_t)DH * NK;
  const float* bp = sbias + (hb * NQ + q0) * NK;

  const Frag aq0 = load_frag(qp, DH);
  const Frag aq1 = load_frag(qp + 32, DH);

  v8f acc[4] = {};
  float rowm[8], rowl[8];
#pragma unroll
  for (int v = 0; v < 8; ++v) { rowm[v] = -3.0e38f; rowl[v] = 0.f; }

  for (int kb = 0; kb < NK; kb += 64) {
    // S = q * k^T  (16 x 64 tile)
    v8f s[4] = {};
#pragma unroll
    for (int j = 0; j < 4; ++j) {
      Frag b0 = load_frag(kp + (size_t)(kb + j * 16) * DH, DH);
      Frag b1 = load_frag(kp + (size_t)(kb + j * 16) * DH + 32, DH);
      s[j] = wmma_bf16(aq0, b0, s[j]);
      s[j] = wmma_bf16(aq1, b1, s[j]);
    }
    // scale + structure bias (NT: 268MB single-use stream, keep out of caches)
#pragma unroll
    for (int j = 0; j < 4; ++j)
#pragma unroll
      for (int v = 0; v < 8; ++v)
        s[j][v] = s[j][v] * 0.03125f +
                  __builtin_nontemporal_load(&bp[(size_t)(mh + v) * NK + kb + j * 16 + cn]);

    // online softmax: each row lives in one 16-lane half (xor<16 stays inside)
#pragma unroll
    for (int v = 0; v < 8; ++v) {
      float mx = fmaxf(fmaxf(s[0][v], s[1][v]), fmaxf(s[2][v], s[3][v]));
#pragma unroll
      for (int off = 1; off < 16; off <<= 1) mx = fmaxf(mx, __shfl_xor(mx, off, 32));
      const float mnew = fmaxf(rowm[v], mx);
      const float alpha = __expf(rowm[v] - mnew);
      rowm[v] = mnew;
      float rs = 0.f;
#pragma unroll
      for (int j = 0; j < 4; ++j) {
        const float p = __expf(s[j][v] - mnew);
        s[j][v] = p;
        rs += p;
      }
#pragma unroll
      for (int off = 1; off < 16; off <<= 1) rs += __shfl_xor(rs, off, 32);
      rowl[v] = rowl[v] * alpha + rs;
#pragma unroll
      for (int nj = 0; nj < 4; ++nj) acc[nj][v] *= alpha;
    }

    // transpose P (C-layout: lane=key) -> A-layout (lane=row) via LDS
#pragma unroll
    for (int j = 0; j < 4; ++j)
#pragma unroll
      for (int v = 0; v < 8; ++v)
        Plds[(mh + v) * 66 + j * 16 + cn] = f2bf(s[j][v]);
    __syncthreads();
    const Frag pa0 = load_frag((const unsigned short*)Plds, 66);
    const Frag pa1 = load_frag((const unsigned short*)Plds + 32, 66);
#pragma unroll
    for (int nj = 0; nj < 4; ++nj) {
      Frag vb0 = load_frag(vp + (size_t)(nj * 16) * NK + kb, NK);
      Frag vb1 = load_frag(vp + (size_t)(nj * 16) * NK + kb + 32, NK);
      acc[nj] = wmma_bf16(pa0, vb0, acc[nj]);
      acc[nj] = wmma_bf16(pa1, vb1, acc[nj]);
    }
    __syncthreads();
  }

  // normalize, residual on q, merge heads
#pragma unroll
  for (int nj = 0; nj < 4; ++nj) {
    const int d = nj * 16 + cn;
#pragma unroll
    for (int v = 0; v < 8; ++v) {
      const int m = mh + v;
      float o = acc[nj][v] / rowl[v];
      o += qf[(hb * NQ + q0 + m) * DH + d];
      OH[((size_t)b * NQ + q0 + m) * DV + h * DH + d] = o;
    }
  }
}

// ----------------------------------------------------------------- layernorm
__global__ __launch_bounds__(256)
void ln_kernel(const float* __restrict__ in, const float* __restrict__ g,
               const float* __restrict__ be, float* __restrict__ outf,
               unsigned short* __restrict__ outb) {
  __shared__ float red[16];
  const int row = blockIdx.x;
  const int t = threadIdx.x;
  const float4 x = ((const float4*)(in + (size_t)row * DV))[t];
  float s  = x.x + x.y + x.z + x.w;
  float ss = x.x * x.x + x.y * x.y + x.z * x.z + x.w * x.w;
#pragma unroll
  for (int off = 1; off < 32; off <<= 1) {
    s  += __shfl_xor(s, off, 32);
    ss += __shfl_xor(ss, off, 32);
  }
  if ((t & 31) == 0) { red[t >> 5] = s; red[8 + (t >> 5)] = ss; }
  __syncthreads();
  float ts = 0.f, tss = 0.f;
#pragma unroll
  for (int w = 0; w < 8; ++w) { ts += red[w]; tss += red[8 + w]; }
  const float mean = ts * (1.f / DV);
  const float var  = tss * (1.f / DV) - mean * mean;
  const float rstd = rsqrtf(var + 1e-5f);
  const float4 gv = ((const float4*)g)[t];
  const float4 bv = ((const float4*)be)[t];
  float4 y;
  y.x = (x.x - mean) * rstd * gv.x + bv.x;
  y.y = (x.y - mean) * rstd * gv.y + bv.y;
  y.z = (x.z - mean) * rstd * gv.z + bv.z;
  y.w = (x.w - mean) * rstd * gv.w + bv.w;
  ((float4*)(outf + (size_t)row * DV))[t] = y;
  if (outb)
    ((uint2*)(outb + (size_t)row * DV))[t] = make_uint2(pack_bf2(y.x, y.y), pack_bf2(y.z, y.w));
}

// ------------------------------------------- output proj + ReLU + residual
// Z = X + relu(X @ Wo^T + bo); grid (32,8), block 256. Same TDM pipeline.
__global__ __launch_bounds__(256)
void out_gemm_kernel(const unsigned short* __restrict__ Xb,
                     const unsigned short* __restrict__ Wob,
                     const float* __restrict__ bo,
                     const float* __restrict__ X,
                     float* __restrict__ Z) {
  __shared__ __align__(16) unsigned short ldsB[2 * WBUFSZ];
  const int wave  = threadIdx.x >> 5;
  const int lane  = threadIdx.x & 31;
  const int cn    = lane & 15;
  const int mh    = (lane >> 4) << 3;
  const int mtile = blockIdx.x * 128 + wave * 16;
  const int ntile = blockIdx.y * 128;
  const bool issuer = (wave == 0);

  v8f acc[8] = {};
  const unsigned ldsbase = (unsigned)(size_t)&ldsB[0];
  const int NIT = DV / 32;

  if (issuer) tdm_stage_weights(Wob + (size_t)ntile * DV, ldsbase);

  for (int it = 0; it < NIT; ++it) {
    const int k0 = it * 32;
    if (it + 1 < NIT) {
      if (issuer) {
        tdm_stage_weights(Wob + (size_t)ntile * DV + (k0 + 32),
                          ldsbase + ((it + 1) & 1) * (WBUFSZ * 2));
        __builtin_amdgcn_s_wait_tensorcnt(1);
      }
    } else if (issuer) {
      __builtin_amdgcn_s_wait_tensorcnt(0);
    }
    __syncthreads();

    const unsigned short* bufp = ldsB + (it & 1) * WBUFSZ;
    Frag af = load_frag(Xb + (size_t)mtile * DV + k0, DV);
#pragma unroll
    for (int nb = 0; nb < 8; ++nb) {
      Frag bf = load_frag(bufp + nb * 16 * WPITCH, WPITCH, 18);
      acc[nb] = wmma_bf16(af, bf, acc[nb]);
    }
    __syncthreads();
  }

#pragma unroll
  for (int nb = 0; nb < 8; ++nb) {
    const int n = ntile + nb * 16 + cn;
    const float bias_n = bo[n];
#pragma unroll
    for (int v = 0; v < 8; ++v) {
      const int m = mtile + mh + v;
      const size_t o = (size_t)m * DV + n;
      Z[o] = X[o] + fmaxf(acc[nb][v] + bias_n, 0.f);
    }
  }
}

// ------------------------------------------------------------------- launch
extern "C" void kernel_launch(void* const* d_in, const int* in_sizes, int n_in,
                              void* d_out, int out_size, void* d_ws, size_t ws_size,
                              hipStream_t stream) {
  const float* Q   = (const float*)d_in[0];
  const float* K   = (const float*)d_in[1];
  const float* SB  = (const float*)d_in[2];
  const float* Wq  = (const float*)d_in[3];
  const float* bq  = (const float*)d_in[4];
  const float* Wk  = (const float*)d_in[5];
  const float* bk  = (const float*)d_in[6];
  const float* Wv  = (const float*)d_in[7];
  const float* bv  = (const float*)d_in[8];
  const float* Wo  = (const float*)d_in[9];
  const float* bo  = (const float*)d_in[10];
  const float* g0  = (const float*)d_in[11];
  const float* be0 = (const float*)d_in[12];
  const float* g1  = (const float*)d_in[13];
  const float* be1 = (const float*)d_in[14];

  char* w = (char*)d_ws;
  auto take = [&](size_t bytes) -> void* {
    void* p = (void*)w;
    w += (bytes + 255) & ~(size_t)255;
    return p;
  };
  unsigned short* Qbf  = (unsigned short*)take((size_t)MROWS * DV * 2);
  unsigned short* Kbf  = (unsigned short*)take((size_t)MROWS * DV * 2);
  unsigned short* Wqb  = (unsigned short*)take((size_t)DV * DV * 2);
  unsigned short* Wkb  = (unsigned short*)take((size_t)DV * DV * 2);
  unsigned short* Wvb  = (unsigned short*)take((size_t)DV * DV * 2);
  unsigned short* Wob  = (unsigned short*)take((size_t)DV * DV * 2);
  float*          qF   = (float*)take((size_t)MROWS * DV * 4);
  unsigned short* qB   = (unsigned short*)take((size_t)MROWS * DV * 2);
  unsigned short* kB   = (unsigned short*)take((size_t)MROWS * DV * 2);
  unsigned short* vTB  = (unsigned short*)take((size_t)MROWS * DV * 2);
  float*          OH   = (float*)take((size_t)MROWS * DV * 4);
  float*          X    = (float*)take((size_t)MROWS * DV * 4);
  unsigned short* Xbf  = (unsigned short*)take((size_t)MROWS * DV * 2);
  float*          Zbuf = (float*)take((size_t)MROWS * DV * 4);

  // 1) f32 -> bf16 conversions
  {
    int n4 = MROWS * DV / 4;
    cvt_bf16_kernel<<<(n4 + 255) / 256, 256, 0, stream>>>((const float4*)Q, (uint2*)Qbf, n4);
    cvt_bf16_kernel<<<(n4 + 255) / 256, 256, 0, stream>>>((const float4*)K, (uint2*)Kbf, n4);
    int w4 = DV * DV / 4;
    cvt_bf16_kernel<<<(w4 + 255) / 256, 256, 0, stream>>>((const float4*)Wq, (uint2*)Wqb, w4);
    cvt_bf16_kernel<<<(w4 + 255) / 256, 256, 0, stream>>>((const float4*)Wk, (uint2*)Wkb, w4);
    cvt_bf16_kernel<<<(w4 + 255) / 256, 256, 0, stream>>>((const float4*)Wv, (uint2*)Wvb, w4);
    cvt_bf16_kernel<<<(w4 + 255) / 256, 256, 0, stream>>>((const float4*)Wo, (uint2*)Wob, w4);
  }

  // 2) fused QKV projections (TDM-staged weights)
  qkv_gemm_kernel<<<dim3(MROWS / 128, DV / 128, 3), 256, 0, stream>>>(
      Qbf, Kbf, Wqb, Wkb, Wvb, bq, bk, bv, qF, qB, kB, vTB);

  // 3) attention (flash, per 16-row query tile)
  attn_kernel<<<dim3(NQ / 16, BN, HN), 32, 0, stream>>>(qB, kB, vTB, qF, SB, OH);

  // 4) LN0
  ln_kernel<<<MROWS, 256, 0, stream>>>(OH, g0, be0, X, Xbf);

  // 5) output projection + ReLU + residual (TDM-staged weights)
  out_gemm_kernel<<<dim3(MROWS / 128, DV / 128), 256, 0, stream>>>(Xbf, Wob, bo, X, Zbuf);

  // 6) LN1 -> final output
  ln_kernel<<<MROWS, 256, 0, stream>>>(Zbuf, g1, be1, (float*)d_out, (unsigned short*)nullptr);
}
